// CRF_41566693490787
// MI455X (gfx1250) — matrix operand order repaired and checked by reading, not compiled, not run on hardware
//
#include <hip/hip_runtime.h>
#include <hip/hip_bf16.h>

typedef __attribute__((ext_vector_type(16))) _Float16 v16h;
typedef __attribute__((ext_vector_type(8)))  _Float16 v8h;
typedef __attribute__((ext_vector_type(8)))  float    v8f;
typedef __attribute__((ext_vector_type(2)))  float    v2f;

#define NPIX 4096
#define NCH  5

// ds_swizzle group-of-32 xor butterfly: offset[14:10]=xor, [4:0]=and_mask
#define SWZ(x, imm) \
  __builtin_bit_cast(float, __builtin_amdgcn_ds_swizzle(__builtin_bit_cast(int, (x)), (imm)))

__device__ __forceinline__ float red_max16(float x) {
  x = fmaxf(x, SWZ(x, 0x041F));   // xor 1
  x = fmaxf(x, SWZ(x, 0x081F));   // xor 2
  x = fmaxf(x, SWZ(x, 0x101F));   // xor 4
  x = fmaxf(x, SWZ(x, 0x201F));   // xor 8
  return x;
}
__device__ __forceinline__ float red_sum16(float x) {
  x += SWZ(x, 0x041F);
  x += SWZ(x, 0x081F);
  x += SWZ(x, 0x101F);
  x += SWZ(x, 0x201F);
  return x;
}

// ---------------------------------------------------------------------------
// Kernel F: per-pixel feature vectors (8-padded, f32) + squared norms.
//   F[n] = [y/5, x/5, 0, 0, 2r0, 2r1, 2r2, 0]
//   k=0..3 dot  -> Gaussian (spatial) kernel distance term
//   k=0..7 dot  -> bilateral kernel distance term
// ---------------------------------------------------------------------------
__global__ void __launch_bounds__(256) crf_features(
    const float* __restrict__ ref, float* __restrict__ F,
    float* __restrict__ sqg, float* __restrict__ sqb) {
  int n = blockIdx.x * blockDim.x + threadIdx.x;   // 0..4095
  float y  = (float)(n >> 6) * 0.2f;
  float x  = (float)(n & 63) * 0.2f;
  float r0 = 2.0f * ref[n];
  float r1 = 2.0f * ref[NPIX + n];
  float r2 = 2.0f * ref[2 * NPIX + n];
  float* f = F + (size_t)n * 8;
  f[0] = y;  f[1] = x;  f[2] = 0.f; f[3] = 0.f;
  f[4] = r0; f[5] = r1; f[6] = r2;  f[7] = 0.f;
  float sg = y * y + x * x;
  sqg[n] = sg;
  sqb[n] = sg + r0 * r0 + r1 * r1 + r2 * r2;
}

// ---------------------------------------------------------------------------
// Kernel A: Ksum via Gram-matrix WMMA (v_wmma_f32_16x16x4_f32).
// One wave per 16x16 tile (i_base, j_base):
//   accg = F_i(k0..3) x F_j(k0..3)^T          (spatial dot)
//   accb = F_i(k4..7) x F_j(k4..7)^T + accg   (full bilateral dot)
//   d2*  = sq*[i] + sq*[j] - 2*acc*           (clamped >= 0)
//   Ksum = exp(-0.5 d2b) + exp(-0.5 d2g), stored f16 (row-contiguous stores).
// f32 A 16x4 layout: lane L<16 -> row M=L, v0/v1 = K 0,1; L>=16 -> K 2,3.
// f32 B 4x16 layout: lane L<16 -> col N=L, v0/v1 = K 0,1; L>=16 -> K 2,3.
// ---------------------------------------------------------------------------
__global__ void __launch_bounds__(256) crf_build_K_wmma(
    const float* __restrict__ F, const float* __restrict__ sqg,
    const float* __restrict__ sqb, _Float16* __restrict__ Kd) {
  const int wid    = blockIdx.x * (blockDim.x >> 5) + (threadIdx.x >> 5); // tile id
  const int lane   = threadIdx.x & 31;
  const int half   = lane >> 4;
  const int lane16 = lane & 15;
  const int i_base = (wid >> 8) << 4;
  const int j_base = (wid & 255) << 4;

  const int ia = i_base + lane16;
  const int jb = j_base + lane16;
  v2f a_lo = *(const v2f*)(F + (size_t)ia * 8 + 2 * half);       // k = 0..3
  v2f a_hi = *(const v2f*)(F + (size_t)ia * 8 + 4 + 2 * half);   // k = 4..7
  v2f b_lo = *(const v2f*)(F + (size_t)jb * 8 + 2 * half);
  v2f b_hi = *(const v2f*)(F + (size_t)jb * 8 + 4 + 2 * half);

  v8f zero = {};
  v8f accg = __builtin_amdgcn_wmma_f32_16x16x4_f32(
      false, a_lo, false, b_lo, (short)0, zero, false, false);
  v8f accb = __builtin_amdgcn_wmma_f32_16x16x4_f32(
      false, a_hi, false, b_hi, (short)0, accg, false, false);

  const float sgj = sqg[jb], sbj = sqb[jb];
#pragma unroll
  for (int v = 0; v < 8; ++v) {
    const int i = i_base + v + 8 * half;
    float d2g = fmaxf(sqg[i] + sgj - 2.0f * accg[v], 0.0f);
    float d2b = fmaxf(sqb[i] + sbj - 2.0f * accb[v], 0.0f);
    Kd[(size_t)i * NPIX + j_base + lane16] =
        (_Float16)(__expf(-0.5f * d2b) + __expf(-0.5f * d2g));
  }
}

// ---------------------------------------------------------------------------
// Kernel A2: out0 = softmax(logits) over channels, f32 [5][4096].
// ---------------------------------------------------------------------------
__global__ void __launch_bounds__(256) crf_init_out(
    const float* __restrict__ logits, float* __restrict__ out) {
  int n = blockIdx.x * blockDim.x + threadIdx.x;
  float z[NCH], zm = -1e30f;
#pragma unroll
  for (int c = 0; c < NCH; ++c) { z[c] = logits[c * NPIX + n]; zm = fmaxf(zm, z[c]); }
  float s = 0.f;
#pragma unroll
  for (int c = 0; c < NCH; ++c) { z[c] = __expf(z[c] - zm); s += z[c]; }
  float inv = 1.0f / s;
#pragma unroll
  for (int c = 0; c < NCH; ++c) out[c * NPIX + n] = z[c] * inv;
}

// ---------------------------------------------------------------------------
// Kernel B: premix compatibility into the WMMA B operand:
//   outM[d][m] = sum_c M[c][d] * out[c][m]   (d<5; d=5..15 zero-pad), f16.
// ---------------------------------------------------------------------------
__global__ void __launch_bounds__(256) crf_mix(
    const float* __restrict__ out, const float* __restrict__ Mmat,
    _Float16* __restrict__ outM) {
  int t = blockIdx.x * blockDim.x + threadIdx.x;    // 0 .. 16*4096-1
  int d = t >> 12;
  int m = t & (NPIX - 1);
  float v = 0.f;
  if (d < NCH) {
#pragma unroll
    for (int c = 0; c < NCH; ++c) v += Mmat[c * NCH + d] * out[c * NPIX + m];
  }
  outM[(size_t)d * NPIX + m] = (_Float16)v;
}

// ---------------------------------------------------------------------------
// Kernel C: msg_mixed = Ksum @ outM^T via v_wmma_f32_16x16x32_f16, fused
// softmax update. 2 waves per block split the m-range; each wave accumulates
// TWO 16-row n-tiles sharing one B load (2 independent WMMA chains). Partial
// sums exchanged through LDS + barrier; ds_swizzle softmax reductions.
// ---------------------------------------------------------------------------
__global__ void __launch_bounds__(64) crf_gemm_update(
    const _Float16* __restrict__ Kd, const _Float16* __restrict__ outM,
    const float* __restrict__ logits, float* __restrict__ dst) {
  __shared__ float xbuf[2][32][8];     // donated partial accumulator per wave

  const int tid    = threadIdx.x;
  const int wave   = tid >> 5;         // 0/1 -> m half
  const int lane   = tid & 31;
  const int half   = lane >> 4;
  const int lane16 = lane & 15;
  const int n_base = blockIdx.x * 32;  // two 16-row tiles per block

  const int m_lo = wave * 2048;
  const _Float16* Arow0 = Kd + (size_t)(n_base + lane16) * NPIX + half * 8 + m_lo;
  const _Float16* Arow1 = Arow0 + (size_t)16 * NPIX;
  const _Float16* Brow  = outM + (size_t)lane16 * NPIX + half * 8 + m_lo;

  v8f acc0 = {}, acc1 = {};
#pragma unroll 4
  for (int m = 0; m < 2048; m += 32) {
    v8h a0lo = *(const v8h*)(Arow0 + m);
    v8h a0hi = *(const v8h*)(Arow0 + m + 16);
    v8h a1lo = *(const v8h*)(Arow1 + m);
    v8h a1hi = *(const v8h*)(Arow1 + m + 16);
    v8h blo = *(const v8h*)(Brow + m);
    v8h bhi = *(const v8h*)(Brow + m + 16);
    v16h b;
#pragma unroll
    for (int k = 0; k < 8; ++k) { b[k] = blo[k]; b[k + 8] = bhi[k]; }
    v16h a0, a1;
#pragma unroll
    for (int k = 0; k < 8; ++k) {
      a0[k] = a0lo[k]; a0[k + 8] = a0hi[k];
      a1[k] = a1lo[k]; a1[k + 8] = a1hi[k];
    }
    acc0 = __builtin_amdgcn_wmma_f32_16x16x32_f16(
        false, a0, false, b, (short)0, acc0, false, false);
    acc1 = __builtin_amdgcn_wmma_f32_16x16x32_f16(
        false, a1, false, b, (short)0, acc1, false, false);
  }

  // Donate the tile the other wave finalizes: wave0 gives acc1, wave1 gives acc0.
#pragma unroll
  for (int v = 0; v < 8; ++v)
    xbuf[wave][lane][v] = wave ? acc0[v] : acc1[v];
  __syncthreads();

  v8f mine = wave ? acc1 : acc0;
#pragma unroll
  for (int v = 0; v < 8; ++v)
    mine[v] += xbuf[1 - wave][lane][v];

  // Epilogue: out[d][n] = softmax_d(logits[d][n] + 3.0 * msg_mixed[d][n]).
  const int tbase = n_base + wave * 16;
  const int d = lane16;
#pragma unroll
  for (int v = 0; v < 8; ++v) {
    const int n = tbase + v + 8 * half;
    float z = (d < NCH) ? (logits[d * NPIX + n] + 3.0f * mine[v]) : -1e30f;
    float zmax = red_max16(z);
    float p = __expf(z - zmax);
    float ps = red_sum16(p);
    if (d < NCH) dst[d * NPIX + n] = p / ps;
  }
}

// ---------------------------------------------------------------------------
// Launch: features -> WMMA Gram-matrix K build -> 5 x (mix -> WMMA GEMM+update)
// ws layout: [0,32MB) Ksum f16 | outM f16 128KB | out f32 128KB |
//            F f32 128KB | sqg 16KB | sqb 16KB
// ---------------------------------------------------------------------------
extern "C" void kernel_launch(void* const* d_in, const int* in_sizes, int n_in,
                              void* d_out, int out_size, void* d_ws, size_t ws_size,
                              hipStream_t stream) {
  const float* logits = (const float*)d_in[0];   // [1,5,64,64] f32
  const float* ref    = (const float*)d_in[1];   // [1,3,64,64] f32
  const float* Mmat   = (const float*)d_in[2];   // [5,5] f32

  char* ws = (char*)d_ws;
  const size_t K_BYTES = (size_t)NPIX * NPIX * 2;          // 32 MB
  _Float16* Kd     = (_Float16*)ws;
  _Float16* outM   = (_Float16*)(ws + K_BYTES);
  float*    outbuf = (float*)(ws + K_BYTES + 131072);
  float*    F      = (float*)(ws + K_BYTES + 131072 + 131072);
  float*    sqg    = (float*)(ws + K_BYTES + 131072 + 131072 + 131072);
  float*    sqb    = (float*)(ws + K_BYTES + 131072 + 131072 + 131072 + 16384);
  float*    final_out = (float*)d_out;

  crf_features<<<NPIX / 256, 256, 0, stream>>>(ref, F, sqg, sqb);
  // 65536 tiles, 8 waves (tiles) per 256-thread block
  crf_build_K_wmma<<<65536 / 8, 256, 0, stream>>>(F, sqg, sqb, Kd);
  crf_init_out<<<NPIX / 256, 256, 0, stream>>>(logits, outbuf);

  for (int it = 0; it < 5; ++it) {
    crf_mix<<<(16 * NPIX) / 256, 256, 0, stream>>>(outbuf, Mmat, outM);
    crf_gemm_update<<<NPIX / 32, 64, 0, stream>>>(
        Kd, outM, logits, (it == 4) ? final_out : outbuf);
  }
}
